// Transformer_89000312308136
// MI455X (gfx1250) — matrix operand order repaired and checked by reading, not compiled, not run on hardware
//
#include <hip/hip_runtime.h>
#include <math.h>

// Problem constants (must match reference)
#define TBATCH 2
#define TS     2048   // seq
#define TD     512    // d_model
#define THD    8      // heads
#define TDH    64     // d_head
#define TF     2048   // d_ff
#define TL     2      // layers

typedef __attribute__((ext_vector_type(16))) _Float16 v16h;
typedef __attribute__((ext_vector_type(8)))  _Float16 v8h;
typedef __attribute__((ext_vector_type(4)))  _Float16 v4h;
typedef __attribute__((ext_vector_type(8)))  float    v8f;

// ---------------- WMMA GEMM: C(MxN) = A(MxK) * B(KxN) [+C] [+bias] [relu] ---
// Block: 128 threads = 4 wave32, computes 64x64 of C.
// Wave w: (w>>1, w&1) -> 32x32 region; 4 accumulators; 2 A frags x 2 B frags.
// LDS: A row-major [64][40] f16 (lane frag = two contiguous 16B runs);
//      B k-innermost per column [64 cols][40 k] f16 (lane frag = two contiguous 16B runs).
#define BM 64
#define BN 64
#define BK 32
#define ALD 40   // row stride in halves (80 B: keeps 16B alignment, skews banks)

__global__ __launch_bounds__(128)
void gemm_wmma(float* __restrict__ C, const float* __restrict__ A,
               const float* __restrict__ Bm, const float* __restrict__ bias,
               int M, int N, int K, int lda, int ldb, int ldc,
               int transB, int doRelu, int beta)
{
  __shared__ __align__(16) _Float16 As[BM * ALD];
  __shared__ __align__(16) _Float16 Bs[BN * ALD];

  const int tid  = threadIdx.x;
  const int bm   = blockIdx.y * BM;
  const int bn   = blockIdx.x * BN;
  const int lane = tid & 31;
  const int w    = tid >> 5;        // 0..3
  const int wm   = (w >> 1) * 32;   // 0 or 32
  const int wn   = (w & 1) * 32;    // 0 or 32
  const int hsel = lane >> 4;       // half-wave select
  const int lrow = lane & 15;

  v8f acc00 = {}, acc01 = {}, acc10 = {}, acc11 = {};

  for (int k0 = 0; k0 < K; k0 += BK) {
    // ---- stage A tile (64x32): float4 global loads -> packed b64 LDS stores
    #pragma unroll
    for (int i = 0; i < 4; ++i) {
      int e = tid + i * 128;        // 0..511 float4 groups
      int r = e >> 3, c4 = (e & 7) * 4;
      float4 f = *(const float4*)&A[(size_t)(bm + r) * lda + (k0 + c4)];
      v4h hx = { (_Float16)f.x, (_Float16)f.y, (_Float16)f.z, (_Float16)f.w };
      *(v4h*)&As[r * ALD + c4] = hx;
    }
    // ---- stage B tile (32x64) into k-innermost layout Bs[col][k]
    if (transB) {                   // B[n*ldb + k]: contiguous k both sides
      #pragma unroll
      for (int i = 0; i < 4; ++i) {
        int e = tid + i * 128;
        int n = e >> 3, c4 = (e & 7) * 4;
        float4 f = *(const float4*)&Bm[(size_t)(bn + n) * ldb + (k0 + c4)];
        v4h hx = { (_Float16)f.x, (_Float16)f.y, (_Float16)f.z, (_Float16)f.w };
        *(v4h*)&Bs[n * ALD + c4] = hx;
      }
    } else {                        // B[k*ldb + n]: float4 along n, scatter k
      #pragma unroll
      for (int i = 0; i < 4; ++i) {
        int e = tid + i * 128;
        int k = e >> 4, n4 = (e & 15) * 4;
        float4 f = *(const float4*)&Bm[(size_t)(k0 + k) * ldb + (bn + n4)];
        Bs[(n4 + 0) * ALD + k] = (_Float16)f.x;
        Bs[(n4 + 1) * ALD + k] = (_Float16)f.y;
        Bs[(n4 + 2) * ALD + k] = (_Float16)f.z;
        Bs[(n4 + 3) * ALD + k] = (_Float16)f.w;
      }
    }
    // prefetch next K tile (gfx1250 global_prefetch_b8, speculative)
    if (k0 + BK < K) {
      __builtin_prefetch(&A[(size_t)(bm + (tid & 63)) * lda + (k0 + BK)], 0, 1);
      const float* pf = transB ? &Bm[(size_t)(bn + (tid & 63)) * ldb + (k0 + BK)]
                               : &Bm[(size_t)(k0 + BK + (tid & 31)) * ldb + bn];
      __builtin_prefetch(pf, 0, 1);
    }
    __syncthreads();

    // ---- A fragments: lane row=lrow(+16), halves = k[h8..h8+8) ++ k[h8+16..h8+24)
    const _Float16* a0p = &As[(wm + lrow) * ALD + hsel * 8];
    const _Float16* a1p = &As[(wm + 16 + lrow) * ALD + hsel * 8];
    v16h a0 = __builtin_shufflevector(*(const v8h*)a0p, *(const v8h*)(a0p + 16),
                                      0,1,2,3,4,5,6,7,8,9,10,11,12,13,14,15);
    v16h a1 = __builtin_shufflevector(*(const v8h*)a1p, *(const v8h*)(a1p + 16),
                                      0,1,2,3,4,5,6,7,8,9,10,11,12,13,14,15);
    // ---- B fragments: lane col=lrow, halves = k[h8..h8+8) ++ k[h8+16..h8+24)
    const _Float16* b0p = &Bs[(wn + lrow) * ALD + hsel * 8];
    const _Float16* b1p = &Bs[(wn + 16 + lrow) * ALD + hsel * 8];
    v16h b0 = __builtin_shufflevector(*(const v8h*)b0p, *(const v8h*)(b0p + 16),
                                      0,1,2,3,4,5,6,7,8,9,10,11,12,13,14,15);
    v16h b1 = __builtin_shufflevector(*(const v8h*)b1p, *(const v8h*)(b1p + 16),
                                      0,1,2,3,4,5,6,7,8,9,10,11,12,13,14,15);

    acc00 = __builtin_amdgcn_wmma_f32_16x16x32_f16(false, a0, false, b0,
                                                   (short)0, acc00, false, false);
    acc01 = __builtin_amdgcn_wmma_f32_16x16x32_f16(false, a0, false, b1,
                                                   (short)0, acc01, false, false);
    acc10 = __builtin_amdgcn_wmma_f32_16x16x32_f16(false, a1, false, b0,
                                                   (short)0, acc10, false, false);
    acc11 = __builtin_amdgcn_wmma_f32_16x16x32_f16(false, a1, false, b1,
                                                   (short)0, acc11, false, false);
    __syncthreads();
  }

  // ---- store: f32 C/D layout: VGPR r -> M = r (lanes0-15) / 8+r (lanes16-31)
  auto store_tile = [&](const v8f& acc, int mo, int no) {
    const int colg = bn + no + lrow;
    #pragma unroll
    for (int r = 0; r < 8; ++r) {
      int ro = hsel ? (8 + r) : r;
      size_t idx = (size_t)(bm + mo + ro) * ldc + colg;
      float v = acc[r];
      if (beta)   v += C[idx];
      if (bias)   v += bias[colg];
      if (doRelu) v = fmaxf(v, 0.0f);
      C[idx] = v;
    }
  };
  store_tile(acc00, wm,      wn);
  store_tile(acc01, wm,      wn + 16);
  store_tile(acc10, wm + 16, wn);
  store_tile(acc11, wm + 16, wn + 16);
}

// ---------------- embedding * sqrt(D) + sinusoidal positional encoding ------
__global__ __launch_bounds__(256)
void embed_pe(float* __restrict__ x, const int* __restrict__ tokens,
              const float* __restrict__ emb)
{
  int idx = blockIdx.x * 256 + threadIdx.x;   // < B*S*D
  int d   = idx & (TD - 1);
  int bs  = idx >> 9;                          // / 512
  int s   = bs & (TS - 1);
  int tok = tokens[bs];
  float pos  = (float)s;
  float freq = __expf(-(float)(d & ~1) * (9.210340371976184f / 512.0f));
  float pe   = (d & 1) ? __cosf(pos * freq) : __sinf(pos * freq);
  x[idx] = emb[(size_t)tok * TD + d] * 22.627416997969522f + pe;
}

// -------- fused skew + causal mask + softmax (one row per block, in-place) --
// P[i,j] = softmax_j( (qk[i,j] + qe[i, S-1-i+j]) * scale ), j<=i else 0
__global__ __launch_bounds__(256)
void softmax_skew(float* __restrict__ P, const float* __restrict__ QE, float scale)
{
  __shared__ float red[256];
  const int i = blockIdx.x;
  const int t = threadIdx.x;
  float* __restrict__ prow = P + (size_t)i * TS;
  const float* __restrict__ qrow = QE + (size_t)i * TS;

  float mx = -3.4e38f;
  for (int j = t; j <= i; j += 256) {
    float v = (prow[j] + qrow[TS - 1 - i + j]) * scale;
    mx = fmaxf(mx, v);
  }
  red[t] = mx; __syncthreads();
  for (int s = 128; s > 0; s >>= 1) { if (t < s) red[t] = fmaxf(red[t], red[t + s]); __syncthreads(); }
  mx = red[0]; __syncthreads();

  float sum = 0.0f;
  for (int j = t; j <= i; j += 256) {
    float v = (prow[j] + qrow[TS - 1 - i + j]) * scale;
    sum += __expf(v - mx);
  }
  red[t] = sum; __syncthreads();
  for (int s = 128; s > 0; s >>= 1) { if (t < s) red[t] += red[t + s]; __syncthreads(); }
  float inv = 1.0f / red[0]; __syncthreads();

  for (int j = t; j < TS; j += 256) {
    float o = 0.0f;
    if (j <= i) {
      float v = (prow[j] + qrow[TS - 1 - i + j]) * scale;
      o = __expf(v - mx) * inv;
    }
    prow[j] = o;
  }
}

// -------- Ap[i,d] = P[i, i-d] for d<=i else 0 -------------------------------
__global__ __launch_bounds__(256)
void ap_gather(float* __restrict__ Ap, const float* __restrict__ P)
{
  int idx = blockIdx.x * 256 + threadIdx.x;   // < S*S
  int i = idx >> 11;                          // / 2048
  int d = idx & (TS - 1);
  Ap[idx] = (d <= i) ? P[((size_t)i << 11) + (i - d)] : 0.0f;
}

// -------- dst = LayerNorm(x (+ z)) * sc + bi (one row of 512 per block) -----
__global__ __launch_bounds__(256)
void ln_res(float* __restrict__ dst, const float* __restrict__ x,
            const float* __restrict__ z, const float* __restrict__ sc,
            const float* __restrict__ bi)
{
  __shared__ float red[256];
  const int row = blockIdx.x;
  const int t   = threadIdx.x;
  const size_t base = (size_t)row * TD;
  float v0 = x[base + t]       + (z ? z[base + t]       : 0.0f);
  float v1 = x[base + t + 256] + (z ? z[base + t + 256] : 0.0f);

  red[t] = v0 + v1; __syncthreads();
  for (int s = 128; s > 0; s >>= 1) { if (t < s) red[t] += red[t + s]; __syncthreads(); }
  float mean = red[0] * (1.0f / TD); __syncthreads();

  float d0 = v0 - mean, d1 = v1 - mean;
  red[t] = d0 * d0 + d1 * d1; __syncthreads();
  for (int s = 128; s > 0; s >>= 1) { if (t < s) red[t] += red[t + s]; __syncthreads(); }
  float rstd = rsqrtf(red[0] * (1.0f / TD) + 1e-5f);

  dst[base + t]       = d0 * rstd * sc[t]       + bi[t];
  dst[base + t + 256] = d1 * rstd * sc[t + 256] + bi[t + 256];
}

// ---------------------------------------------------------------------------
extern "C" void kernel_launch(void* const* d_in, const int* in_sizes, int n_in,
                              void* d_out, int out_size, void* d_ws, size_t ws_size,
                              hipStream_t stream)
{
  (void)in_sizes; (void)n_in; (void)out_size; (void)ws_size;

  const int*   tokens = (const int*)  d_in[0];
  const float* emb    = (const float*)d_in[1];
  const float* Wq     = (const float*)d_in[2];
  const float* bq     = (const float*)d_in[3];
  const float* Wk     = (const float*)d_in[4];
  const float* bk     = (const float*)d_in[5];
  const float* Wv     = (const float*)d_in[6];
  const float* bv     = (const float*)d_in[7];
  const float* rel    = (const float*)d_in[8];
  const float* vrel   = (const float*)d_in[9];
  const float* W1     = (const float*)d_in[10];
  const float* b1     = (const float*)d_in[11];
  const float* W2     = (const float*)d_in[12];
  const float* b2     = (const float*)d_in[13];
  const float* ln1s   = (const float*)d_in[14];
  const float* ln1b   = (const float*)d_in[15];
  const float* ln2s   = (const float*)d_in[16];
  const float* ln2b   = (const float*)d_in[17];
  const float* lnfs   = (const float*)d_in[18];
  const float* lnfb   = (const float*)d_in[19];
  float* out = (float*)d_out;

  // workspace layout (floats): ~109 MB total
  const size_t NX = (size_t)TBATCH * TS * TD;      // 2,097,152
  float* ws = (float*)d_ws;
  float* x  = ws;
  float* Qb = x  + NX;
  float* Kb = Qb + NX;
  float* Vb = Kb + NX;
  float* zb = Vb + NX;
  float* hb = zb + NX;                             // B*S*F
  float* qk = hb + (size_t)TBATCH * TS * TF;       // S*S
  float* qe = qk + (size_t)TS * TS;                // S*S

  auto gemm = [&](float* C, const float* A, const float* Bmat, const float* bias,
                  int M, int N, int K, int lda, int ldb, int ldc,
                  int transB, int relu, int beta) {
    gemm_wmma<<<dim3(N / BN, M / BM), 128, 0, stream>>>(
        C, A, Bmat, bias, M, N, K, lda, ldb, ldc, transB, relu, beta);
  };

  const int Mrows = TBATCH * TS;  // 4096
  const float scale = 0.125f;     // 1/sqrt(64)

  embed_pe<<<(unsigned)(NX / 256), 256, 0, stream>>>(x, tokens, emb);

  for (int l = 0; l < TL; ++l) {
    const float* wq = Wq + (size_t)l * TD * TD;
    const float* wk = Wk + (size_t)l * TD * TD;
    const float* wv = Wv + (size_t)l * TD * TD;
    const float* er = rel  + (size_t)l * TS * TDH;
    const float* ev = vrel + (size_t)l * TS * TDH;

    // Q/K/V projections (stored (B*S, D); head h = column slice [h*64, h*64+64))
    gemm(Qb, x, wq, bq + l * TD, Mrows, TD, TD, TD, TD, TD, 0, 0, 0);
    gemm(Kb, x, wk, bk + l * TD, Mrows, TD, TD, TD, TD, TD, 0, 0, 0);
    gemm(Vb, x, wv, bv + l * TD, Mrows, TD, TD, TD, TD, TD, 0, 0, 0);

    for (int b = 0; b < TBATCH; ++b) {
      for (int h = 0; h < THD; ++h) {
        const float* Qbh = Qb + (size_t)b * TS * TD + h * TDH;
        const float* Kbh = Kb + (size_t)b * TS * TD + h * TDH;
        const float* Vbh = Vb + (size_t)b * TS * TD + h * TDH;
        float*       Zbh = zb + (size_t)b * TS * TD + h * TDH;

        // qk = Q * K^T   (S x S, K=64)
        gemm(qk, Qbh, Kbh, nullptr, TS, TS, TDH, TD, TD, TS, 1, 0, 0);
        // qe = Q * er^T  (S x S, K=64)
        gemm(qe, Qbh, er, nullptr, TS, TS, TDH, TD, TDH, TS, 1, 0, 0);
        // P = softmax over (qk + skewed qe) with causal mask (in-place in qk)
        softmax_skew<<<TS, 256, 0, stream>>>(qk, qe, scale);
        // Z = P * V      (S x 64, K=S), write into z at head column offset
        gemm(Zbh, qk, Vbh, nullptr, TS, TDH, TS, TS, TD, TD, 0, 0, 0);
        // Ap[i,d] = P[i, i-d]  (into qe buffer)
        ap_gather<<<(TS * TS) / 256, 256, 0, stream>>>(qe, qk);
        // Z += Ap * vrel (S x 64, K=S), accumulate
        gemm(Zbh, qe, ev, nullptr, TS, TDH, TS, TS, TDH, TD, 0, 0, 1);
      }
    }

    // x = LN(x + z)
    ln_res<<<Mrows, 256, 0, stream>>>(x, x, zb, ln1s + l * TD, ln1b + l * TD);
    // h = relu(x*W1 + b1); ffn = h*W2 + b2 (reuse zb)
    gemm(hb, x, W1 + (size_t)l * TD * TF, b1 + l * TF, Mrows, TF, TD, TD, TF, TF, 0, 1, 0);
    gemm(zb, hb, W2 + (size_t)l * TF * TD, b2 + l * TD, Mrows, TD, TF, TF, TD, TD, 0, 0, 0);
    // x = LN(x + ffn)
    ln_res<<<Mrows, 256, 0, stream>>>(x, x, zb, ln2s + l * TD, ln2b + l * TD);
  }

  // final LayerNorm -> d_out (fp32)
  ln_res<<<Mrows, 256, 0, stream>>>(out, x, nullptr, lnfs, lnfb);
}
